// Generator_44555990728950
// MI455X (gfx1250) — matrix operand order repaired and checked by reading, not compiled, not run on hardware
//
#include <hip/hip_runtime.h>
#include <hip/hip_bf16.h>

typedef __bf16 bf16_t;
typedef __attribute__((ext_vector_type(16))) __bf16 v16bf;
typedef __attribute__((ext_vector_type(8)))  float  v8f;
typedef unsigned int v4u __attribute__((ext_vector_type(4)));
typedef int          v8i __attribute__((ext_vector_type(8)));
typedef int          v4i __attribute__((ext_vector_type(4)));

#define Hdim   128
#define NN     32
#define LAYERS 16
#define A_CLS  10
#define HYBC   3
#define FNODE  17
#define EFEAT  4
#define NEDGE  62            // 2*(NN-1)
#define BTOT   4096

#define GM      8            // molecules per workgroup
#define ROWS    (GM*NN)      // 256
#define THREADS 256
#define SX      132          // padded LDS row stride (bf16 elems)
#define SW      132

// flat output offsets (return-order concat)
#define OFF_NF    0
#define OFF_ALP   (BTOT*NN*FNODE)                  // 2228224
#define OFF_HLP   (OFF_ALP + BTOT)                 // 2232320
#define OFF_EATTR (OFF_HLP + BTOT)                 // 2236416
#define OFF_MASK  (OFF_EATTR + BTOT*NEDGE*EFEAT)   // 3252224

#define HAS_TDM __has_builtin(__builtin_amdgcn_tensor_load_to_lds)

__constant__ float c_maxval[A_CLS] = {4.f,3.f,2.f,1.f,4.f,2.f,6.f,1.f,4.f,4.f};

__device__ __forceinline__ unsigned hash3(unsigned a, unsigned b, unsigned c) {
  unsigned h = a*0x9E3779B1u ^ b*0x85EBCA77u ^ c*0xC2B2AE3Du;
  h ^= h >> 16; h *= 0x7FEB352Du; h ^= h >> 15; h *= 0x846CA68Bu; h ^= h >> 16;
  return h;
}
__device__ __forceinline__ float u01(unsigned h) {
  return (float)(h >> 8) * 5.9604645e-08f + 1.0e-7f;
}
__device__ __forceinline__ float gumbelf(unsigned h) {
  float u = u01(h);
  return -__logf(-__logf(u));
}
__device__ __forceinline__ float sigmoidf(float x) { return 1.f/(1.f+__expf(-x)); }

// ---- 16x32 bf16 A-fragment from LDS (row-major, stride s); ISA 7.12.2 layout ----
__device__ __forceinline__ v16bf load_afrag(const bf16_t* X, int mrow, int kb,
                                            int lr, int half, int s) {
  v16bf a;
  const bf16_t* row = X + (mrow + lr)*s;
#pragma unroll
  for (int p = 0; p < 8; ++p) {
    int k = kb + ((p&3)<<1) + (half<<3) + ((p>=4)?16:0);
    a[2*p]   = row[k];
    a[2*p+1] = row[k+1];
  }
  return a;
}
// ---- 32x16 bf16 B-fragment from LDS holding W^T (row n = out col) ----
__device__ __forceinline__ v16bf load_bfrag(const bf16_t* Wt, int ncol, int kb,
                                            int lr, int half, int s) {
  v16bf b;
  const bf16_t* row = Wt + (ncol + lr)*s + kb + (half<<3);
#pragma unroll
  for (int e = 0; e < 16; ++e) b[e] = row[(e < 8) ? e : e + 8];
  return b;
}

#if HAS_TDM
// Issue a TDM 2D tile load: 128x128 f32, global (row-major, stride 128) -> LDS.
__device__ __forceinline__ void tdm_load_128x128_f32(const float* gsrc,
                                                     unsigned lds_byte_addr) {
  unsigned long long ga = (unsigned long long)(const void*)gsrc;
  v4u g0;
  g0.x = 1u;                                            // count=1, user mode
  g0.y = lds_byte_addr;                                 // lds_addr
  g0.z = (unsigned)ga;                                  // global_addr[31:0]
  g0.w = (unsigned)((ga >> 32) & 0x1FFFFFFull) | (2u << 30);  // addr[56:32] | type=2
  v8i g1;
  g1[0] = 0x20000;                 // wg_mask=0, data_size=2 (4B)
  g1[1] = (int)(128u << 16);       // tensor_dim0 = 128
  g1[2] = (int)(128u << 16);       // tensor_dim1 = 128
  g1[3] = (int)(128u << 16);       // tile_dim0   = 128
  g1[4] = 128;                     // tile_dim1   = 128, tile_dim2=0
  g1[5] = 128;                     // tensor_dim0_stride = 128
  g1[6] = 0; g1[7] = 0;
  v4i g2 = {0,0,0,0}, g3 = {0,0,0,0};
#if defined(__clang_major__) && __clang_major__ >= 23
  v8i g4 = {0,0,0,0,0,0,0,0};
  __builtin_amdgcn_tensor_load_to_lds(g0, g1, g2, g3, g4, 0);
#else
  __builtin_amdgcn_tensor_load_to_lds(g0, g1, g2, g3, 0);
#endif
}
#endif

// ============================ kernel 1: h0 ============================
__global__ __launch_bounds__(THREADS)
void k_h0(const float* __restrict__ noise, const float* __restrict__ w1,
          const float* __restrict__ b1, bf16_t* __restrict__ h0out) {
  __shared__ bf16_t As[128*SX];
  __shared__ bf16_t Bs[128*SW];
  __shared__ float  bias[128];
  int t = threadIdx.x;
  int row0 = blockIdx.x * 128;

  for (int idx = t; idx < 128*128; idx += THREADS) {
    int m = idx >> 7, k = idx & 127;
    As[m*SX+k] = (bf16_t)noise[(size_t)(row0+m)*128 + k];
  }
  for (int idx = t; idx < 128*128; idx += THREADS) {
    int k = idx >> 7, n = idx & 127;
    Bs[n*SW+k] = (bf16_t)w1[k*128+n];          // W^T
  }
  if (t < 128) bias[t] = b1[t];
  __syncthreads();

  int wave = t >> 5, lane = t & 31, half = lane >> 4, lr = lane & 15;
  int mrow = wave << 4;                         // one row-tile per wave
  v16bf a0 = load_afrag(As, mrow, 0,  lr, half, SX);
  v16bf a1 = load_afrag(As, mrow, 32, lr, half, SX);
  v16bf a2 = load_afrag(As, mrow, 64, lr, half, SX);
  v16bf a3 = load_afrag(As, mrow, 96, lr, half, SX);
  for (int ct = 0; ct < 8; ++ct) {
    int ncol = ct << 4;
    v8f acc = {};
    acc = __builtin_amdgcn_wmma_f32_16x16x32_bf16(false, a0, false,
            load_bfrag(Bs, ncol, 0,  lr, half, SW), (short)0, acc, false, false);
    acc = __builtin_amdgcn_wmma_f32_16x16x32_bf16(false, a1, false,
            load_bfrag(Bs, ncol, 32, lr, half, SW), (short)0, acc, false, false);
    acc = __builtin_amdgcn_wmma_f32_16x16x32_bf16(false, a2, false,
            load_bfrag(Bs, ncol, 64, lr, half, SW), (short)0, acc, false, false);
    acc = __builtin_amdgcn_wmma_f32_16x16x32_bf16(false, a3, false,
            load_bfrag(Bs, ncol, 96, lr, half, SW), (short)0, acc, false, false);
    int n = ncol + lr;
#pragma unroll
    for (int r = 0; r < 8; ++r) {
      int m = mrow + r + (half<<3);
      float v = acc[r] + bias[n];
      h0out[(size_t)(row0+m)*128 + n] = (bf16_t)(v > 0.f ? v : 0.f);
    }
  }
}

// ===== head-phase LDS layout, aliased over the TDM f32 staging buffer =====
struct HeadLds {
  float nf[ROWS][FNODE+1];
  float sw_atom[128*A_CLS], sb_atom[A_CLS];
  float sw_hyb[128*HYBC],   sb_hyb[HYBC];
  float sw_deg[128], sw_chg[128], sw_arom[128], sb3[3];
  float sw_eex[2*FNODE], sb_eex;
  float sw_ety[2*FNODE*EFEAT], sb_ety[EFEAT];
  float mol_alp[GM], mol_hlp[GM];
};

// ===================== kernel 2: 16 GAT layers + heads =====================
__global__ __launch_bounds__(THREADS)
void k_gat(const bf16_t* __restrict__ h0,
           const float* __restrict__ gat_w, const float* __restrict__ gat_b,
           const float* __restrict__ att_src, const float* __restrict__ att_dst,
           const float* __restrict__ w_atom, const float* __restrict__ b_atom,
           const float* __restrict__ w_hyb,  const float* __restrict__ b_hyb,
           const float* __restrict__ w_deg,  const float* __restrict__ b_deg,
           const float* __restrict__ w_chg,  const float* __restrict__ b_chg,
           const float* __restrict__ w_arom, const float* __restrict__ b_arom,
           const float* __restrict__ w_eex,  const float* __restrict__ b_eex,
           const float* __restrict__ w_ety,  const float* __restrict__ b_ety,
           float* __restrict__ out) {
  __shared__ bf16_t Xs[ROWS*SX];
  __shared__ bf16_t Hs[ROWS*SX];
  __shared__ bf16_t Ws[128*SW];
  __shared__ float  bias[128], asv[128], adv[128];
  __shared__ float  esrc[ROWS], edst[ROWS];
  __shared__ __align__(16) unsigned char uscratch[128*128*4];  // 64 KB
  float*   Fw = (float*)uscratch;            // TDM f32 staging (layer phase)
  HeadLds* HD = (HeadLds*)uscratch;          // head phase (barrier-separated)

  int t = threadIdx.x;
  int mol0 = blockIdx.x * GM;
  int wave = t >> 5, lane = t & 31, half = lane >> 4, lr = lane & 15;

  // ---- load x: row t = node (t&31) of molecule (t>>5); broadcast of h0 row ----
  {
    const unsigned* s32 = (const unsigned*)(h0 + (size_t)(mol0 + (t >> 5))*128);
    unsigned* d32 = (unsigned*)(Xs + t*SX);
#pragma unroll 8
    for (int j = 0; j < 64; ++j) d32[j] = s32[j];
  }

  // ---- 16 fused residual GAT layers ----
  for (int l = 0; l < LAYERS; ++l) {
    const float* Wl = gat_w + l*128*128;
    __syncthreads();                          // prev-iter consumers done
#if HAS_TDM
    if (wave == 0) {                          // TDM: EXEC-independent DMA
      tdm_load_128x128_f32(Wl, (unsigned)(size_t)Fw);
      __builtin_amdgcn_s_wait_tensorcnt(0);
    }
    __syncthreads();                          // Fw visible to all waves
    for (int idx = t; idx < 128*128; idx += THREADS) {
      int k = idx >> 7, n = idx & 127;
      Ws[n*SW+k] = (bf16_t)Fw[idx];           // stage W^T in bf16 from LDS
    }
#else
    for (int idx = t; idx < 128*128; idx += THREADS) {
      int k = idx >> 7, n = idx & 127;
      Ws[n*SW+k] = (bf16_t)Wl[idx];           // stage W^T in bf16 from global
    }
#endif
    if (t < 128) {
      bias[t] = gat_b[l*128+t];
      asv[t]  = att_src[l*128+t];
      adv[t]  = att_dst[l*128+t];
    }
    if (l+1 < LAYERS)                          // warm L2 for next layer
      __builtin_prefetch((const void*)(gat_w + (l+1)*16384 + t*64), 0, 1);
    __syncthreads();

    // h = x @ W + b : 2 row-tiles per wave, A-fragments cached in registers
    for (int rti = 0; rti < 2; ++rti) {
      int mrow = ((wave<<1) + rti) << 4;
      v16bf a0 = load_afrag(Xs, mrow, 0,  lr, half, SX);
      v16bf a1 = load_afrag(Xs, mrow, 32, lr, half, SX);
      v16bf a2 = load_afrag(Xs, mrow, 64, lr, half, SX);
      v16bf a3 = load_afrag(Xs, mrow, 96, lr, half, SX);
      for (int ct = 0; ct < 8; ++ct) {
        int ncol = ct << 4;
        v8f acc = {};
        acc = __builtin_amdgcn_wmma_f32_16x16x32_bf16(false, a0, false,
                load_bfrag(Ws, ncol, 0,  lr, half, SW), (short)0, acc, false, false);
        acc = __builtin_amdgcn_wmma_f32_16x16x32_bf16(false, a1, false,
                load_bfrag(Ws, ncol, 32, lr, half, SW), (short)0, acc, false, false);
        acc = __builtin_amdgcn_wmma_f32_16x16x32_bf16(false, a2, false,
                load_bfrag(Ws, ncol, 64, lr, half, SW), (short)0, acc, false, false);
        acc = __builtin_amdgcn_wmma_f32_16x16x32_bf16(false, a3, false,
                load_bfrag(Ws, ncol, 96, lr, half, SW), (short)0, acc, false, false);
        int n = ncol + lr;
#pragma unroll
        for (int r = 0; r < 8; ++r) {
          int m = mrow + r + (half<<3);
          Hs[m*SX+n] = (bf16_t)(acc[r] + bias[n]);
        }
      }
    }
    __syncthreads();

    // attention scores e_src/e_dst (one row per thread)
    {
      float es = 0.f, ed = 0.f;
      const bf16_t* hr = Hs + t*SX;
#pragma unroll 16
      for (int k = 0; k < 128; ++k) {
        float hv = (float)hr[k];
        es += hv * asv[k];
        ed += hv * adv[k];
      }
      esrc[t] = es; edst[t] = ed;
    }
    __syncthreads();

    // tridiagonal masked softmax + aggregate + residual ReLU (in place on Xs)
    {
      int i = t & 31;
      float d = edst[t];
      float s0 = (i > 0)    ? (d + esrc[t-1]) : -1e30f;
      float s1 =              (d + esrc[t]);
      float s2 = (i < NN-1) ? (d + esrc[t+1]) : -1e30f;
      s0 = s0 > 0.f ? s0 : 0.2f*s0;            // leaky_relu(0.2)
      s1 = s1 > 0.f ? s1 : 0.2f*s1;
      s2 = s2 > 0.f ? s2 : 0.2f*s2;
      float mx = fmaxf(s1, fmaxf(s0, s2));
      float e0 = __expf(s0-mx), e1 = __expf(s1-mx), e2 = __expf(s2-mx);
      float inv = 1.f/(e0+e1+e2);
      e0 *= inv; e1 *= inv; e2 *= inv;         // invalid neighbors -> exactly 0
      int r0 = (i > 0)    ? t-1 : t;
      int r2 = (i < NN-1) ? t+1 : t;
      const bf16_t* h0r = Hs + r0*SX;
      const bf16_t* h1r = Hs + t*SX;
      const bf16_t* h2r = Hs + r2*SX;
      bf16_t* xr = Xs + t*SX;
#pragma unroll 8
      for (int k = 0; k < 128; ++k) {
        float o = e0*(float)h0r[k] + e1*(float)h1r[k] + e2*(float)h2r[k]
                + (float)xr[k];
        xr[k] = (bf16_t)(o > 0.f ? o : 0.f);
      }
    }
  }
  __syncthreads();                             // layer phase done; Fw dead

  // ---- stage head weights (into scratch union) ----
  for (int idx = t; idx < 128*A_CLS; idx += THREADS) HD->sw_atom[idx] = w_atom[idx];
  for (int idx = t; idx < 128*HYBC;  idx += THREADS) HD->sw_hyb[idx]  = w_hyb[idx];
  if (t < 128) { HD->sw_deg[t]=w_deg[t]; HD->sw_chg[t]=w_chg[t]; HD->sw_arom[t]=w_arom[t]; }
  if (t < A_CLS) HD->sb_atom[t] = b_atom[t];
  if (t < HYBC)  HD->sb_hyb[t]  = b_hyb[t];
  if (t == 0) { HD->sb3[0]=b_deg[0]; HD->sb3[1]=b_chg[0]; HD->sb3[2]=b_arom[0];
                HD->sb_eex = b_eex[0]; }
  if (t < 2*FNODE) HD->sw_eex[t] = w_eex[t];
  for (int idx = t; idx < 2*FNODE*EFEAT; idx += THREADS) HD->sw_ety[idx] = w_ety[idx];
  if (t < EFEAT) HD->sb_ety[t] = b_ety[t];
  if (t < GM) { HD->mol_alp[t] = 0.f; HD->mol_hlp[t] = 0.f; }
  __syncthreads();

  // ---- node heads (one node per thread) ----
  {
    int i = t & 31, g = t >> 5;
    int b = mol0 + g;
    float lg[A_CLS], lh[HYBC];
#pragma unroll
    for (int a = 0; a < A_CLS; ++a) lg[a] = HD->sb_atom[a];
#pragma unroll
    for (int a = 0; a < HYBC;  ++a) lh[a] = HD->sb_hyb[a];
    float sd = HD->sb3[0], sg = HD->sb3[1], sa = HD->sb3[2];
    const bf16_t* xr = Xs + t*SX;
    for (int k = 0; k < 128; ++k) {
      float xv = (float)xr[k];
#pragma unroll
      for (int a = 0; a < A_CLS; ++a) lg[a] += xv * HD->sw_atom[k*A_CLS + a];
#pragma unroll
      for (int a = 0; a < HYBC;  ++a) lh[a] += xv * HD->sw_hyb[k*HYBC + a];
      sd += xv * HD->sw_deg[k]; sg += xv * HD->sw_chg[k]; sa += xv * HD->sw_arom[k];
    }
    // atom head: log_softmax + Gumbel-max sample
    float mx = lg[0];
#pragma unroll
    for (int a = 1; a < A_CLS; ++a) mx = fmaxf(mx, lg[a]);
    float se = 0.f;
#pragma unroll
    for (int a = 0; a < A_CLS; ++a) se += __expf(lg[a]-mx);
    float lse = mx + __logf(se);
    int aidx = 0; float best = -1e30f, sel = 0.f;
#pragma unroll
    for (int a = 0; a < A_CLS; ++a) {
      float v = lg[a] + gumbelf(hash3((unsigned)b, (unsigned)i, 0x100u + a));
      if (v > best) { best = v; aidx = a; sel = lg[a]; }
    }
    atomicAdd(&HD->mol_alp[g], (sel - lse) * (1.f/(float)NN));
    // hyb head
    float hmx = fmaxf(lh[0], fmaxf(lh[1], lh[2]));
    float hse = __expf(lh[0]-hmx) + __expf(lh[1]-hmx) + __expf(lh[2]-hmx);
    float hlse = hmx + __logf(hse);
    int hidx = 0; float hbest = -1e30f, hsel = 0.f;
#pragma unroll
    for (int a = 0; a < HYBC; ++a) {
      float v = lh[a] + gumbelf(hash3((unsigned)b, (unsigned)i, 0x200u + a));
      if (v > hbest) { hbest = v; hidx = a; hsel = lh[a]; }
    }
    atomicAdd(&HD->mol_hlp[g], (hsel - hlse) * (1.f/(float)NN));
    // scalar heads
    float deg  = sigmoidf(sd);
    float chg  = tanhf(sg);
    float ap   = sigmoidf(sa);
    float arom = (u01(hash3((unsigned)b, (unsigned)i, 0x300u)) < ap) ? 1.f : 0.f;
    float val  = c_maxval[aidx] * 0.2f;
    // node_features = [onehot10, deg, chg, onehot3, arom, val]
    float* od = out + OFF_NF + ((size_t)b*NN + i)*FNODE;
    float* fl = HD->nf[t];
#pragma unroll
    for (int q = 0; q < A_CLS; ++q) { float v = (q==aidx)?1.f:0.f; od[q]=v; fl[q]=v; }
    od[10] = deg;  fl[10] = deg;
    od[11] = chg;  fl[11] = chg;
#pragma unroll
    for (int q = 0; q < HYBC; ++q) { float v = (q==hidx)?1.f:0.f; od[12+q]=v; fl[12+q]=v; }
    od[15] = arom; fl[15] = arom;
    od[16] = val;  fl[16] = val;
  }
  __syncthreads();

  if (t < GM) {
    out[OFF_ALP + mol0 + t] = HD->mol_alp[t];
    out[OFF_HLP + mol0 + t] = HD->mol_hlp[t];
  }

  // ---- edge heads: 62 chain edges (both directions) per molecule ----
  for (int eidx = t; eidx < GM*NEDGE; eidx += THREADS) {
    int g = eidx / NEDGE, e = eidx % NEDGE;
    int u, v;
    if (e < NN-1) { u = e; v = e+1; } else { u = e-(NN-1)+1; v = e-(NN-1); }
    const float* fu = HD->nf[(g<<5)+u];
    const float* fv = HD->nf[(g<<5)+v];
    float ex = HD->sb_eex;
    float tl[EFEAT];
#pragma unroll
    for (int c = 0; c < EFEAT; ++c) tl[c] = HD->sb_ety[c];
#pragma unroll
    for (int q = 0; q < FNODE; ++q) {
      float a1 = fu[q], a2 = fv[q];
      ex += a1*HD->sw_eex[q] + a2*HD->sw_eex[FNODE+q];
#pragma unroll
      for (int c = 0; c < EFEAT; ++c)
        tl[c] += a1*HD->sw_ety[q*EFEAT+c] + a2*HD->sw_ety[(FNODE+q)*EFEAT+c];
    }
    int b = mol0 + g;
    float p = sigmoidf(ex);
    out[OFF_MASK + (size_t)b*NEDGE + e] = (p > 0.5f) ? 1.f : 0.f;
    int ti = 0; float best = -1e30f;
#pragma unroll
    for (int c = 0; c < EFEAT; ++c) {
      float vz = tl[c] + gumbelf(hash3((unsigned)b, 0x400u + e, c));
      if (vz > best) { best = vz; ti = c; }
    }
    float* oe = out + OFF_EATTR + ((size_t)b*NEDGE + e)*EFEAT;
#pragma unroll
    for (int c = 0; c < EFEAT; ++c) oe[c] = (c==ti)?1.f:0.f;
  }
}

extern "C" void kernel_launch(void* const* d_in, const int* in_sizes, int n_in,
                              void* d_out, int out_size, void* d_ws, size_t ws_size,
                              hipStream_t stream) {
  (void)in_sizes; (void)n_in; (void)out_size; (void)ws_size;
  const float* noise   = (const float*)d_in[0];
  const float* w1      = (const float*)d_in[1];
  const float* b1      = (const float*)d_in[2];
  const float* gat_w   = (const float*)d_in[3];
  const float* gat_b   = (const float*)d_in[4];
  const float* att_src = (const float*)d_in[5];
  const float* att_dst = (const float*)d_in[6];
  const float* w_atom  = (const float*)d_in[7];
  const float* b_atom  = (const float*)d_in[8];
  const float* w_hyb   = (const float*)d_in[9];
  const float* b_hyb   = (const float*)d_in[10];
  const float* w_deg   = (const float*)d_in[11];
  const float* b_deg   = (const float*)d_in[12];
  const float* w_chg   = (const float*)d_in[13];
  const float* b_chg   = (const float*)d_in[14];
  const float* w_arom  = (const float*)d_in[15];
  const float* b_arom  = (const float*)d_in[16];
  const float* w_eex   = (const float*)d_in[17];
  const float* b_eex   = (const float*)d_in[18];
  const float* w_ety   = (const float*)d_in[19];
  const float* b_ety   = (const float*)d_in[20];

  bf16_t* h0 = (bf16_t*)d_ws;                 // [4096][128] bf16 = 1 MB

  k_h0<<<BTOT/128, THREADS, 0, stream>>>(noise, w1, b1, h0);
  k_gat<<<BTOT/GM, THREADS, 0, stream>>>(h0, gat_w, gat_b, att_src, att_dst,
      w_atom, b_atom, w_hyb, b_hyb, w_deg, b_deg, w_chg, b_chg,
      w_arom, b_arom, w_eex, b_eex, w_ety, b_ety, (float*)d_out);
}
